// MultiHeadAttention_2044404433122
// MI455X (gfx1250) — matrix operand order repaired and checked by reading, not compiled, not run on hardware
//
#include <hip/hip_runtime.h>
#include <hip/hip_bf16.h>
#include <stdint.h>

// ---------------------------------------------------------------------------
// Types matching gfx1250 WMMA builtin signatures
// ---------------------------------------------------------------------------
typedef __attribute__((ext_vector_type(16))) __bf16 v16bf;
typedef __attribute__((ext_vector_type(8)))  __bf16 v8bf;
typedef __attribute__((ext_vector_type(8)))  float  v8f;

#define DEV static __device__ __forceinline__

// Problem constants (reference: T=2048, B=2, H=16, d=128)
constexpr int T_LEN   = 2048;
constexpr int BATCH   = 2;
constexpr int NHEAD   = 16;
constexpr int HDIM    = 128;
constexpr int MODEL   = NHEAD * HDIM;   // 2048
constexpr int ROWS    = T_LEN * BATCH;  // 4096
constexpr int N3      = 3 * MODEL;      // 6144
constexpr int ROT     = 64;
constexpr int POS_OFF = 8192 - T_LEN;   // rope [-T:] slice offset

// Workspace layout (bytes)
constexpr size_t OFF_XB   = 0;
constexpr size_t SZ_XB    = (size_t)ROWS * MODEL * 2;   // x in bf16
constexpr size_t OFF_WQKV = OFF_XB + SZ_XB;
constexpr size_t SZ_WQKV  = (size_t)N3 * MODEL * 2;     // in_proj_weight bf16
constexpr size_t OFF_QKV  = OFF_WQKV + SZ_WQKV;
constexpr size_t SZ_QKV   = (size_t)ROWS * N3 * 2;      // qkv (post-bias) bf16
constexpr size_t OFF_Q    = OFF_QKV + SZ_QKV;
constexpr size_t SZ_HEADS = (size_t)ROWS * MODEL * 2;
constexpr size_t OFF_K    = OFF_Q  + SZ_HEADS;
constexpr size_t OFF_VT   = OFF_K  + SZ_HEADS;          // V stored (bh, d, T)
constexpr size_t OFF_WOT  = OFF_VT + SZ_HEADS;
constexpr size_t SZ_WOT   = (size_t)MODEL * MODEL * 2;  // out_w transposed bf16
constexpr size_t OFF_AO   = OFF_WOT + SZ_WOT;           // attention output bf16

// ---------------------------------------------------------------------------
// Fragment helpers (CDNA5 bf16 WMMA VGPR layouts, wave32)
// ---------------------------------------------------------------------------
DEV v16bf frag_cat(v8bf lo, v8bf hi) {
  union { v16bf v; v8bf h[2]; } u;
  u.h[0] = lo; u.h[1] = hi;
  return u.v;
}

// A-matrix 16x32: lane L<16 holds row L, K = [0..7],[16..23]; lane L+16: [8..15],[24..31]
// hi8 = (lane>=16) ? 8 : 0
DEV v16bf load_fragA(const __bf16* row, int kb, int hi8) {
  const v8bf* p0 = (const v8bf*)(row + kb + hi8);
  const v8bf* p1 = (const v8bf*)(row + kb + 16 + hi8);
  return frag_cat(*p0, *p1);
}

// B-matrix 32x16: lane n<16 holds col n, K = [0..7],[16..23]; lane n+16: [8..15],[24..31]
// hi16 = (lane>=16) ? 16 : 0
DEV v16bf load_fragB(const __bf16* row, int kb, int hi16) {
  const v8bf* p0 = (const v8bf*)(row + kb + hi16 / 2);
  const v8bf* p1 = (const v8bf*)(row + kb + 16 + hi16 / 2);
  return frag_cat(*p0, *p1);
}

DEV v8f wmma_bf16(v16bf a, v16bf b, v8f c) {
  return __builtin_amdgcn_wmma_f32_16x16x32_bf16(false, a, false, b,
                                                 (short)0, c, false, false);
}

// ---------------------------------------------------------------------------
// 16-lane butterfly reductions via DPP row ops (pure VALU — co-executes with
// the XDL WMMA pipe; avoids ds_bpermute + s_wait_dscnt round-trips).
// xor1 = quad_perm(1,0,3,2)=0xB1, xor2 = quad_perm(2,3,0,1)=0x4E,
// xor4 = row_half_mirror=0x141,   xor8 = row_mirror=0x140 (row = 16 lanes).
// ---------------------------------------------------------------------------
template <int CTRL>
DEV float dpp16(float v) {
  int x = __float_as_int(v);
  return __int_as_float(__builtin_amdgcn_update_dpp(x, x, CTRL, 0xf, 0xf, true));
}
DEV float grp16_max(float v) {
  v = fmaxf(v, dpp16<0xB1>(v));
  v = fmaxf(v, dpp16<0x4E>(v));
  v = fmaxf(v, dpp16<0x141>(v));
  v = fmaxf(v, dpp16<0x140>(v));
  return v;
}
DEV float grp16_sum(float v) {
  v += dpp16<0xB1>(v);
  v += dpp16<0x4E>(v);
  v += dpp16<0x141>(v);
  v += dpp16<0x140>(v);
  return v;
}

// ---------------------------------------------------------------------------
// Elementwise conversion kernels
// ---------------------------------------------------------------------------
__global__ void __launch_bounds__(256)
cvt_f32_bf16(const float* __restrict__ src, __bf16* __restrict__ dst, int n) {
  int tid = blockIdx.x * blockDim.x + threadIdx.x;
  if (tid < n) dst[tid] = (__bf16)src[tid];
}

// woT[m][k] = out_w[k][m]  (makes out_w columns K-contiguous for NT GEMM)
__global__ void __launch_bounds__(256)
cvt_transpose(const float* __restrict__ src, __bf16* __restrict__ dst) {
  int tid = blockIdx.x * blockDim.x + threadIdx.x;
  int m = tid >> 11;              // / MODEL
  int k = tid & (MODEL - 1);
  dst[(size_t)m * MODEL + k] = (__bf16)src[(size_t)k * MODEL + m];
}

// ---------------------------------------------------------------------------
// NT GEMM:  C[r, c] = sum_k A[r,k] * B[c,k] + bias[c]
// Block: 256 threads = 8 waves; wave tile 32x32 (2x2 WMMA acc); block tile 64x128.
// ---------------------------------------------------------------------------
template <bool OUT_BF16>
__global__ void __launch_bounds__(256, 1)
gemm_nt_bf16(const __bf16* __restrict__ A, const __bf16* __restrict__ Bm,
             const float* __restrict__ bias, void* __restrict__ Cout,
             int Ncols, int Kd) {
  const int lane = threadIdx.x & 31;
  const int wave = threadIdx.x >> 5;
  const int l15  = lane & 15;
  const int hi   = lane >> 4;
  const int hi8  = hi * 8, hi16 = hi * 16;

  const int m0 = blockIdx.x * 64  + (wave & 1) * 32;
  const int n0 = blockIdx.y * 128 + (wave >> 1) * 32;

  const __bf16* a0r = A  + (size_t)(m0 + l15) * Kd;
  const __bf16* a1r = a0r + (size_t)16 * Kd;
  const __bf16* b0r = Bm + (size_t)(n0 + l15) * Kd;
  const __bf16* b1r = b0r + (size_t)16 * Kd;

  v8f acc[2][2] = {};
  for (int kb = 0; kb < Kd; kb += 32) {
    v16bf a0 = load_fragA(a0r, kb, hi8);
    v16bf a1 = load_fragA(a1r, kb, hi8);
    v16bf b0 = load_fragB(b0r, kb, hi16);
    v16bf b1 = load_fragB(b1r, kb, hi16);
    acc[0][0] = wmma_bf16(a0, b0, acc[0][0]);
    acc[0][1] = wmma_bf16(a0, b1, acc[0][1]);
    acc[1][0] = wmma_bf16(a1, b0, acc[1][0]);
    acc[1][1] = wmma_bf16(a1, b1, acc[1][1]);
  }

  // Epilogue: C-fragment layout -> lane holds col l15 (+16 per jn), row j+8*hi
#pragma unroll
  for (int i = 0; i < 2; i++)
#pragma unroll
    for (int jn = 0; jn < 2; jn++) {
      const int col = n0 + jn * 16 + l15;
      const float bv = bias[col];
#pragma unroll
      for (int j = 0; j < 8; j++) {
        const int row = m0 + i * 16 + j + 8 * hi;
        const float v = acc[i][jn][j] + bv;
        const size_t idx = (size_t)row * Ncols + col;
        if constexpr (OUT_BF16) ((__bf16*)Cout)[idx] = (__bf16)v;
        else                    ((float*)Cout)[idx]  = v;
      }
    }
}

// ---------------------------------------------------------------------------
// RoPE (rotate-every-two, first 64 dims, positions offset by 6144) + repack:
//   q -> (b*H+h, T, d) bf16, pre-scaled by 1/sqrt(d) (folds softmax scale)
//   k -> (b*H+h, T, d) bf16
//   v -> transposed (b*H+h, d, T) bf16
// ---------------------------------------------------------------------------
__global__ void __launch_bounds__(256)
rope_pack(const __bf16* __restrict__ qkv,
          __bf16* __restrict__ q, __bf16* __restrict__ k,
          __bf16* __restrict__ vT) {
  const int tid = blockIdx.x * blockDim.x + threadIdx.x; // over T*B*H*HDIM
  const int dim = tid & (HDIM - 1);
  const int h   = (tid >> 7)  & (NHEAD - 1);
  const int b   = (tid >> 11) & (BATCH - 1);
  const int t   = tid >> 12;

  const size_t rb = (size_t)(t * BATCH + b) * N3 + h * HDIM;
  float qv = (float)qkv[rb + dim];
  float kv = (float)qkv[rb + MODEL + dim];
  float vv = (float)qkv[rb + 2 * MODEL + dim];

  if (dim < ROT) {
    const float qp = (float)qkv[rb + (dim ^ 1)];
    const float kp = (float)qkv[rb + MODEL + (dim ^ 1)];
    const int i2 = dim & ~1;
    // inv_freq = 10000^{-i2/64} = exp(-i2 * ln(10000)/64)
    const float inv = __expf(-(float)i2 * (9.210340371976184f / (float)ROT));
    float s, c;
    __sincosf((float)(t + POS_OFF) * inv, &s, &c);
    if (dim & 1) { qv = qv * c + qp * s; kv = kv * c + kp * s; }
    else         { qv = qv * c - qp * s; kv = kv * c - kp * s; }
  }

  const float qscale = 0.08838834764831845f; // 1/sqrt(128), folded into Q
  const size_t hb = (size_t)(b * NHEAD + h);
  q [(hb * T_LEN + t) * HDIM + dim] = (__bf16)(qv * qscale);
  k [(hb * T_LEN + t) * HDIM + dim] = (__bf16)kv;
  vT[(hb * HDIM + dim) * T_LEN + t] = (__bf16)vv;
}

// ---------------------------------------------------------------------------
// FlashAttention-style causal attention.
// Grid: (T/128, B*H); block 256 = 8 waves; each wave owns 16 query rows.
// Per 32-key step: 8 wmma for S = Q K^T, online softmax (DPP row reductions),
// P re-layout via per-wave LDS bounce (+ s_wait_dscnt), 8 wmma for O += P V.
// Key loop split: unmasked main phase + masked diagonal tail (1 step).
// ---------------------------------------------------------------------------
__global__ void __launch_bounds__(256, 1)
attn_fa(const __bf16* __restrict__ Qh, const __bf16* __restrict__ Kh,
        const __bf16* __restrict__ Vt, __bf16* __restrict__ Ao) {
  __shared__ __bf16 pbuf[8][16 * 32];   // 1 KB per wave

  const int lane = threadIdx.x & 31;
  const int wave = threadIdx.x >> 5;
  const int l15  = lane & 15;
  const int hi   = lane >> 4;
  const int hi8  = hi * 8, hi16 = hi * 16;

  const int bh = blockIdx.y;
  const int b  = bh >> 4;
  const int h  = bh & 15;
  const int q0 = (blockIdx.x * 8 + wave) * 16;

  const __bf16* Q = Qh + (size_t)bh * T_LEN * HDIM;
  const __bf16* K = Kh + (size_t)bh * T_LEN * HDIM;
  const __bf16* V = Vt + (size_t)bh * HDIM * T_LEN;

  // Persistent Q fragments: 16 rows x 128 dims = 4 A-fragments (pre-scaled)
  v16bf qf[4];
  const __bf16* qrow = Q + (size_t)(q0 + l15) * HDIM;
#pragma unroll
  for (int i = 0; i < 4; i++) qf[i] = load_fragA(qrow, i * 32, hi8);

  v8f o[8] = {};                         // 16 x 128 f32 accumulator
  float m[8], ls[8];
#pragma unroll
  for (int j = 0; j < 8; j++) { m[j] = -1e30f; ls[j] = 0.0f; }

  __bf16* pw = &pbuf[wave][0];

  auto step = [&](int kc, bool masked) {
    // --- S = Q K^T (16 x 32), two 16x16 tiles ---
    v8f s0 = {}, s1 = {};
#pragma unroll
    for (int kk = 0; kk < 4; kk++) {
      v16bf kb0 = load_fragB(K + (size_t)(kc + l15) * HDIM,      kk * 32, hi16);
      v16bf kb1 = load_fragB(K + (size_t)(kc + 16 + l15) * HDIM, kk * 32, hi16);
      s0 = wmma_bf16(qf[kk], kb0, s0);
      s1 = wmma_bf16(qf[kk], kb1, s1);
    }

    // --- (tail only) causal mask + online softmax via DPP row reductions ---
    const int c0 = kc + l15, c1 = c0 + 16;
#pragma unroll
    for (int j = 0; j < 8; j++) {
      float v0 = s0[j], v1 = s1[j];
      if (masked) {
        const int row = q0 + j + 8 * hi;
        v0 = (c0 <= row) ? v0 : -1e30f;
        v1 = (c1 <= row) ? v1 : -1e30f;
      }
      const float mx = grp16_max(fmaxf(v0, v1));
      const float mn = fmaxf(m[j], mx);
      const float al = __expf(m[j] - mn);
      m[j] = mn;
      const float p0 = __expf(v0 - mn);
      const float p1 = __expf(v1 - mn);
      ls[j] = ls[j] * al + grp16_sum(p0 + p1);
#pragma unroll
      for (int t = 0; t < 8; t++) o[t][j] *= al;
      // C-fragment -> row-major P tile in per-wave LDS (for A-fragment reload)
      __bf16* pr = pw + (size_t)(j + 8 * hi) * 32;
      pr[l15]      = (__bf16)p0;
      pr[l15 + 16] = (__bf16)p1;
    }

    // In-wave DS ordering: drain LDS stores before fragment reload
    asm volatile("s_wait_dscnt 0" ::: "memory");

    v16bf pf;
    {
      const __bf16* prow = pw + (size_t)l15 * 32;
      pf = frag_cat(*(const v8bf*)(prow + hi8),
                    *(const v8bf*)(prow + 16 + hi8));
    }

    // --- O += P V  (V^T rows are K-contiguous per head-dim) ---
#pragma unroll
    for (int t = 0; t < 8; t++) {
      const __bf16* vrow = V + (size_t)(t * 16 + l15) * T_LEN + kc;
      v16bf vf = frag_cat(*(const v8bf*)(vrow + hi8),
                          *(const v8bf*)(vrow + 16 + hi8));
      o[t] = wmma_bf16(pf, vf, o[t]);
    }
  };

  // Unmasked main phase: window fully below the diagonal (kc+31 <= q0)
  int kc = 0;
  for (; kc + 31 <= q0; kc += 32) step(kc, false);
  // Masked diagonal tail
  for (; kc <= q0 + 15; kc += 32) step(kc, true);

  // --- normalize + write (T, B, M) bf16 ---
#pragma unroll
  for (int j = 0; j < 8; j++) {
    const float inv = 1.0f / ls[j];
    const int trow = q0 + j + 8 * hi;
    const size_t base = ((size_t)(trow * BATCH + b)) * MODEL + h * HDIM;
#pragma unroll
    for (int t = 0; t < 8; t++)
      Ao[base + t * 16 + l15] = (__bf16)(o[t][j] * inv);
  }
}

// ---------------------------------------------------------------------------
// Launcher
// ---------------------------------------------------------------------------
extern "C" void kernel_launch(void* const* d_in, const int* in_sizes, int n_in,
                              void* d_out, int out_size, void* d_ws, size_t ws_size,
                              hipStream_t stream) {
  (void)in_sizes; (void)n_in; (void)out_size; (void)ws_size;

  const float* x    = (const float*)d_in[0];
  const float* wqkv = (const float*)d_in[1];
  const float* bqkv = (const float*)d_in[2];
  const float* wout = (const float*)d_in[3];
  const float* bout = (const float*)d_in[4];
  // d_in[5] = attn_mask: triu(-1e10, k=1) == causal; applied analytically.

  char* ws = (char*)d_ws;
  __bf16* xb   = (__bf16*)(ws + OFF_XB);
  __bf16* wqb  = (__bf16*)(ws + OFF_WQKV);
  __bf16* qkvb = (__bf16*)(ws + OFF_QKV);
  __bf16* qh   = (__bf16*)(ws + OFF_Q);
  __bf16* kh   = (__bf16*)(ws + OFF_K);
  __bf16* vt   = (__bf16*)(ws + OFF_VT);
  __bf16* wot  = (__bf16*)(ws + OFF_WOT);
  __bf16* ao   = (__bf16*)(ws + OFF_AO);

  // 1) precision conversion
  cvt_f32_bf16<<<(ROWS * MODEL) / 256, 256, 0, stream>>>(x, xb, ROWS * MODEL);
  cvt_f32_bf16<<<(N3 * MODEL) / 256, 256, 0, stream>>>(wqkv, wqb, N3 * MODEL);
  cvt_transpose<<<(MODEL * MODEL) / 256, 256, 0, stream>>>(wout, wot);

  // 2) QKV projection: qkv = x · Wqkv^T + b   (4096 x 6144)
  gemm_nt_bf16<true><<<dim3(ROWS / 64, N3 / 128), 256, 0, stream>>>(
      xb, wqb, bqkv, qkvb, N3, MODEL);

  // 3) RoPE + head repack (V transposed, Q pre-scaled by 1/sqrt(d))
  rope_pack<<<(ROWS * MODEL) / 256, 256, 0, stream>>>(qkvb, qh, kh, vt);

  // 4) causal flash attention
  attn_fa<<<dim3(T_LEN / 128, BATCH * NHEAD), 256, 0, stream>>>(qh, kh, vt, ao);

  // 5) output projection: out = attn · out_w + out_b   (fp32 to d_out)
  gemm_nt_bf16<false><<<dim3(ROWS / 64, MODEL / 128), 256, 0, stream>>>(
      ao, wot, bout, d_out, MODEL, MODEL);
}
